// CVIB_67216238183228
// MI455X (gfx1250) — matrix operand order repaired and checked by reading, not compiled, not run on hardware
//
#include <hip/hip_runtime.h>
#include <hip/hip_bf16.h>

// ---------------- problem constants (match reference) ----------------
#define N_USERS   100000
#define N_ITEMS   40000
#define N_NODES   140000        // N_USERS + N_ITEMS
#define EMB       128
#define NEDGES    1000000
#define BATCH     8192
#define TWOB      16384         // 2*BATCH, also size of sampled arrays
#define ALPHA_C   0.1f
#define GAMMA_C   0.01f

typedef __attribute__((ext_vector_type(16))) _Float16 v16h;
typedef __attribute__((ext_vector_type(8)))  float    v8f;

// total floats per embedding table, and float4 count
#define NTOT   (N_NODES * EMB)          // 17,920,000
#define NTOT4  (NTOT / 4)               // 4,480,000
#define NU4    ((N_USERS * EMB) / 4)    // 3,200,000

// ---------------------------------------------------------------------
// sums[0] = sum over pos of log(pred) + sum over neg of log(1-pred)
// sums[1] = sum of pred over 2B
// sums[2] = sum of pred*log(pred) over 2B
// sums[3] = sum of pred_ul over 2B
__global__ void zero_sums_kernel(float* sums) {
    int t = threadIdx.x;
    if (t < 4) sums[t] = 0.0f;
}

// cur = acc = concat(user_emb, item_emb), vectorized as float4
__global__ void init_kernel(const float4* __restrict__ ue4,
                            const float4* __restrict__ ie4,
                            float4* __restrict__ cur4,
                            float4* __restrict__ acc4) {
    int i = blockIdx.x * blockDim.x + threadIdx.x;
    if (i >= NTOT4) return;
    float4 v = (i < NU4) ? ue4[i] : ie4[i - NU4];
    cur4[i] = v;
    acc4[i] = v;
}

__global__ void zero_buf_kernel(float4* __restrict__ b4) {
    int i = blockIdx.x * blockDim.x + threadIdx.x;
    if (i >= NTOT4) return;
    b4[i] = make_float4(0.f, 0.f, 0.f, 0.f);
}

__global__ void acc_add_kernel(float4* __restrict__ acc4,
                               const float4* __restrict__ src4) {
    int i = blockIdx.x * blockDim.x + threadIdx.x;
    if (i >= NTOT4) return;
    float4 a = acc4[i];
    float4 s = src4[i];
    a.x += s.x; a.y += s.y; a.z += s.z; a.w += s.w;
    acc4[i] = a;
}

// One wave per edge: handles both directions (symmetric adjacency).
// rows = [edge_user, edge_item+N_USERS], cols = [edge_item+N_USERS, edge_user]
// vals[e]        -> row = user, col = item_node
// vals[e+NEDGES] -> row = item_node, col = user
__global__ void spmm_kernel(const int*   __restrict__ eu,
                            const int*   __restrict__ ei,
                            const float* __restrict__ vals,
                            const float* __restrict__ cur,
                            float*       __restrict__ nxt) {
    int gid  = blockIdx.x * blockDim.x + threadIdx.x;
    int wave = gid >> 5;
    int lane = gid & 31;
    if (wave >= NEDGES) return;

    // light prefetch of upcoming edge metadata (global_prefetch_b8)
    if (wave + 4096 < NEDGES) {
        __builtin_prefetch(&eu[wave + 4096], 0, 1);
        __builtin_prefetch(&ei[wave + 4096], 0, 1);
    }

    int u  = eu[wave];
    int it = ei[wave] + N_USERS;
    float vf = vals[wave];            // row=u,  col=it
    float vb = vals[wave + NEDGES];   // row=it, col=u

    size_t ru = (size_t)u  * EMB + lane * 4;
    size_t ri = (size_t)it * EMB + lane * 4;

    const float4 xi = *(const float4*)(cur + ri);   // cur[it]
    const float4 xu = *(const float4*)(cur + ru);   // cur[u]

    float* du = nxt + ru;
    float* di = nxt + ri;
    atomicAdd(du + 0, vf * xi.x);
    atomicAdd(du + 1, vf * xi.y);
    atomicAdd(du + 2, vf * xi.z);
    atomicAdd(du + 3, vf * xi.w);
    atomicAdd(di + 0, vb * xu.x);
    atomicAdd(di + 1, vb * xu.y);
    atomicAdd(di + 2, vb * xu.z);
    atomicAdd(di + 3, vb * xu.w);
}

// ---------------------------------------------------------------------
// WMMA-based batched dot products: 16 (user,item) pairs per wave.
// D = A x B + C with A = 16x32 f16 (user rows, K-chunk), B = 32x16 f16
// (item rows laid out as columns). The diagonal of the 16x16 f32 result
// is the 16 dot products for this K-chunk; accumulate over K = 128.
// The final /(N_LAYERS+1)=0.25 scale is folded into both f16 conversions.
//
// mode 0: p<BATCH -> (users[p], pos[p]) label=1 ; else (users[p-B], neg[p-B]) label=0
//         accumulates sums[0..2]
// mode 1: (sampled_user[p], sampled_items[p]) -> accumulates sums[3]
__global__ void dots_wmma_kernel(const float* __restrict__ acc,
                                 const int* __restrict__ idx_u,
                                 const int* __restrict__ idx_a,
                                 const int* __restrict__ idx_b,
                                 float* __restrict__ sums,
                                 int mode) {
    int gid  = blockIdx.x * blockDim.x + threadIdx.x;
    int wave = gid >> 5;
    int lane = threadIdx.x & 31;
    int base = wave * 16;            // first pair index of this tile

    int m = lane & 15;               // row (A) / col (B) this lane feeds
    int p = base + m;

    int un, en;
    if (mode == 0) {
        if (p < BATCH) { un = idx_u[p];         en = N_USERS + idx_a[p]; }
        else           { un = idx_u[p - BATCH]; en = N_USERS + idx_b[p - BATCH]; }
    } else {
        un = idx_u[p]; en = N_USERS + idx_a[p];
    }

    const float* urow = acc + (size_t)un * EMB;
    const float* erow = acc + (size_t)en * EMB;

    // A 16x32 f16, contiguous k order chosen by this code (matches B): lanes 0-15 K=h, lanes 16-31 K=16+h.
    // B 32x16 f16, same contiguous k order as A: lanes 0-15 K=h, lanes 16-31 K=16+h.
    int kbaseA = (lane < 16) ? 0 : 8;
    int koffB  = (lane < 16) ? 0 : 16;

    v8f c = {};
    for (int kc = 0; kc < EMB; kc += 32) {
        v16h a, b;
#pragma unroll
        for (int h = 0; h < 16; ++h) {
            int kA = koffB + h;
            a[h] = (_Float16)(0.25f * urow[kc + kA]);
            b[h] = (_Float16)(0.25f * erow[kc + koffB + h]);
        }
        c = __builtin_amdgcn_wmma_f32_16x16x32_f16(
                /*neg_a=*/false, a, /*neg_b=*/false, b,
                /*c_mod=*/(short)0, c, /*reuse_a=*/false, /*reuse_b=*/false);
    }

    // Extract diagonal of C (16x16 f32):
    //   d < 8  -> lane d      , VGPR d
    //   d >= 8 -> lane 16 + d , VGPR d - 8
    float carr[8];
#pragma unroll
    for (int i = 0; i < 8; ++i) carr[i] = c[i];

    bool active = (lane < 8) || (lane >= 24);
    if (active) {
        int d   = (lane < 8) ? lane : (lane - 16);  // diag index within tile
        int sel = (lane < 8) ? lane : (lane - 24);  // VGPR index
        float dot = carr[sel];
        int pp = base + d;

        float e    = expf(-dot);
        float pred = 1.0f / (1.0f + e);
        float logp = -log1pf(e);          // log(sigmoid(dot))
        float log1mp = -dot + logp;       // log(1 - sigmoid(dot))

        if (mode == 0) {
            float contrib = (pp < BATCH) ? logp : log1mp;
            atomicAdd(&sums[0], contrib);
            atomicAdd(&sums[1], pred);
            atomicAdd(&sums[2], pred * logp);
        } else {
            atomicAdd(&sums[3], pred);
        }
    }
}

__global__ void finalize_kernel(const float* __restrict__ sums,
                                float* __restrict__ out) {
    if (threadIdx.x == 0 && blockIdx.x == 0) {
        const float inv = 1.0f / (float)TWOB;
        float bce = -sums[0] * inv;
        float pa  = sums[1] * inv;       // mean(pred)
        float q   = sums[2] * inv;       // mean(pred*log(pred))
        float ul  = sums[3] * inv;       // mean(pred_ul)
        float info = ALPHA_C * (-pa * logf(ul) - (1.0f - pa) * logf(1.0f - ul))
                   + GAMMA_C * q;
        out[0] = bce;
        out[1] = info;
    }
}

// ---------------------------------------------------------------------
extern "C" void kernel_launch(void* const* d_in, const int* in_sizes, int n_in,
                              void* d_out, int out_size, void* d_ws, size_t ws_size,
                              hipStream_t stream) {
    const float* user_emb      = (const float*)d_in[0];
    const float* item_emb      = (const float*)d_in[1];
    const float* graph_vals    = (const float*)d_in[2];
    const int*   edge_user     = (const int*)d_in[3];
    const int*   edge_item     = (const int*)d_in[4];
    const int*   users         = (const int*)d_in[5];
    const int*   pos_items     = (const int*)d_in[6];
    const int*   neg_items     = (const int*)d_in[7];
    const int*   sampled_user  = (const int*)d_in[8];
    const int*   sampled_items = (const int*)d_in[9];
    float* out = (float*)d_out;

    // workspace layout: [sums: 256B][cur: NB][nxt: NB][acc: NB]
    const size_t NB = (size_t)NTOT * sizeof(float);   // 71,680,000 (256-aligned)
    char* ws = (char*)d_ws;
    float* sums = (float*)ws;
    float* bufA = (float*)(ws + 256);
    float* bufB = (float*)(ws + 256 + NB);
    float* acc  = (float*)(ws + 256 + 2 * NB);

    const int TB = 256;
    const int elem_blocks = (NTOT4 + TB - 1) / TB;        // 17,500
    const int spmm_blocks = (NEDGES * 32) / TB;           // 125,000 (exact)
    const int dot_blocks  = (TWOB / 16) * 32 / TB;        // 128 (exact, full waves)

    zero_sums_kernel<<<1, 32, 0, stream>>>(sums);
    init_kernel<<<elem_blocks, TB, 0, stream>>>(
        (const float4*)user_emb, (const float4*)item_emb,
        (float4*)bufA, (float4*)acc);

    float* cur = bufA;
    float* nxt = bufB;
    for (int layer = 0; layer < 3; ++layer) {
        zero_buf_kernel<<<elem_blocks, TB, 0, stream>>>((float4*)nxt);
        spmm_kernel<<<spmm_blocks, TB, 0, stream>>>(
            edge_user, edge_item, graph_vals, cur, nxt);
        acc_add_kernel<<<elem_blocks, TB, 0, stream>>>((float4*)acc, (const float4*)nxt);
        float* t = cur; cur = nxt; nxt = t;
    }

    dots_wmma_kernel<<<dot_blocks, TB, 0, stream>>>(
        acc, users, pos_items, neg_items, sums, /*mode=*/0);
    dots_wmma_kernel<<<dot_blocks, TB, 0, stream>>>(
        acc, sampled_user, sampled_items, nullptr, sums, /*mode=*/1);

    finalize_kernel<<<1, 32, 0, stream>>>(sums, out);

    (void)in_sizes; (void)n_in; (void)out_size; (void)ws_size;
}
